// Model_39676907883638
// MI455X (gfx1250) — compile-verified, matrix operand order checked by reading, not run on hardware
//
#include <hip/hip_runtime.h>
#include <stdint.h>

#define TB 256
#define BM 128
#define BN 128
#define BK 32
#define LDT 40   // padded LDS row stride (bf16 elems) to dodge bank conflicts

typedef __attribute__((ext_vector_type(16))) __bf16 v16bf;
typedef __attribute__((ext_vector_type(2)))  __bf16 v2bf;
typedef __attribute__((ext_vector_type(8)))  float  v8f;
typedef __attribute__((ext_vector_type(4)))  int    v4i;

union FragAB {
  v16bf v;
  uint4 q[2];
};

static constexpr int kB    = 8;
static constexpr int kS    = 1025;
static constexpr int kDQ   = 4096;
static constexpr int kDTOT = 8200;
static constexpr int kDO   = 4096;
static constexpr int kBS   = kB * kS;   // 8200
static constexpr int kLDQK = 8192;      // bf16 row stride of qs|ks buffer

// ---- fp32 -> bf16 via hardware cvt (not integer bit-twiddling) -------------
__device__ __forceinline__ unsigned int pack2bf(float lo, float hi) {
  v2bf p;
  p[0] = (__bf16)lo;
  p[1] = (__bf16)hi;
  unsigned int r;
  __builtin_memcpy(&r, &p, 4);
  return r;
}
__device__ __forceinline__ unsigned short bf1(float x) {
  __bf16 b = (__bf16)x;
  unsigned short r;
  __builtin_memcpy(&r, &b, 2);
  return r;
}

// ---- CDNA5 async global->LDS (ASYNCcnt) with sync fallback -----------------
#if defined(__gfx1250__) && __has_builtin(__builtin_amdgcn_global_load_async_to_lds_b128)
#define USE_ASYNC_LDS 1
#else
#define USE_ASYNC_LDS 0
#endif

__device__ __forceinline__ void async_ld16(const unsigned short* g, unsigned short* l) {
#if USE_ASYNC_LDS
  __builtin_amdgcn_global_load_async_to_lds_b128(
      (__attribute__((address_space(1))) v4i*)(uintptr_t)g,
      (__attribute__((address_space(3))) v4i*)(uintptr_t)l,
      0, 0);
#else
  *(uint4*)l = *(const uint4*)g;
#endif
}

__device__ __forceinline__ void wait_async_lds() {
#if USE_ASYNC_LDS
 #if __has_builtin(__builtin_amdgcn_s_wait_asynccnt)
  __builtin_amdgcn_s_wait_asynccnt(0);
 #else
  asm volatile("s_wait_asynccnt 0x0" ::: "memory");
 #endif
#endif
}

// ---------------------------------------------------------------------------
// Shared bf16 WMMA mainloop: C[128x128] per block, 8 waves in a 4x2 grid,
// each wave owns a 32x64 tile = 2x4 WMMA accumulators. Double-buffered LDS
// fed by async b128 copies; one barrier + one asynccnt wait per K step.
// Rows beyond aRows/bRows are simply not loaded (stale LDS only feeds output
// rows/cols that the epilogue never stores).
// ---------------------------------------------------------------------------
__device__ __forceinline__ void wmma_mainloop(
    const unsigned short* __restrict__ A0,   // &A[mBase][0]
    const unsigned short* __restrict__ B0,   // &B[nBase][0]
    int aRows, int bRows, int lda, int ldb, int K,
    unsigned short* As, unsigned short* Bs,  // each [2 * BM * LDT]
    v8f (&acc)[2][4])
{
  const int tid  = threadIdx.x;
  const int lane = tid & 31;
  const int wave = tid >> 5;
  const int wm = wave >> 1;        // 0..3 along M
  const int wn = wave & 1;         // 0..1 along N
  const int h  = lane >> 4, mn = lane & 15;
  const int bufStride = BM * LDT;

  // loader mapping: thread -> (row, 16-col chunk); 2 b128 asyncs per tile
  const int lrow = tid >> 1;            // 0..127
  const int lcol = (tid & 1) * 16;      // 0 or 16
  const bool aok = lrow < aRows;
  const bool bok = lrow < bRows;
  const unsigned short* ga = A0 + (size_t)lrow * lda + lcol;
  const unsigned short* gb = B0 + (size_t)lrow * ldb + lcol;
  unsigned short* lA = As + lrow * LDT + lcol;
  unsigned short* lB = Bs + lrow * LDT + lcol;

  // prologue: stage K-block 0 into buffer 0
  if (aok) { async_ld16(ga, lA); async_ld16(ga + 8, lA + 8); }
  if (bok) { async_ld16(gb, lB); async_ld16(gb + 8, lB + 8); }

  const int nk = K / BK;
  for (int kb = 0; kb < nk; ++kb) {
    const int cur = kb & 1;
    wait_async_lds();
    __syncthreads();
    if (kb + 1 < nk) {
      const int nxt = 1 - cur;
      const int ko  = (kb + 1) * BK;
      if (aok) { async_ld16(ga + ko,     lA + nxt * bufStride);
                 async_ld16(ga + ko + 8, lA + nxt * bufStride + 8); }
      if (bok) { async_ld16(gb + ko,     lB + nxt * bufStride);
                 async_ld16(gb + ko + 8, lB + nxt * bufStride + 8); }
    }
    const unsigned short* Ab = As + cur * bufStride;
    const unsigned short* Bb = Bs + cur * bufStride;

    FragAB af[2], bfr[4];
    #pragma unroll
    for (int i = 0; i < 2; ++i) {
      const int r = (wm * 32 + i * 16 + mn) * LDT + 8 * h;
      af[i].q[0] = *(const uint4*)&Ab[r];
      af[i].q[1] = *(const uint4*)&Ab[r + 16];
    }
    #pragma unroll
    for (int j = 0; j < 4; ++j) {
      const int r = (wn * 64 + j * 16 + mn) * LDT + 8 * h;
      bfr[j].q[0] = *(const uint4*)&Bb[r];
      bfr[j].q[1] = *(const uint4*)&Bb[r + 16];
    }
    #pragma unroll
    for (int i = 0; i < 2; ++i)
      #pragma unroll
      for (int j = 0; j < 4; ++j)
        acc[i][j] = __builtin_amdgcn_wmma_f32_16x16x32_bf16(
            false, af[i].v, false, bfr[j].v, (short)0, acc[i][j], false, false);
  }
}

// ---------------------------------------------------------------------------
// fp32 -> bf16 streaming convert (one pass; makes GEMM operands L2-resident)
// ---------------------------------------------------------------------------
__global__ void __launch_bounds__(TB)
k_cvt_bf16(const float* __restrict__ src, unsigned short* __restrict__ dst, int n4)
{
  for (int i = blockIdx.x * TB + threadIdx.x; i < n4; i += gridDim.x * TB) {
    const float4 f = *(const float4*)(src + (size_t)i * 4);
    uint2 p;
    p.x = pack2bf(f.x, f.y);
    p.y = pack2bf(f.z, f.w);
    *(uint2*)(dst + (size_t)i * 4) = p;
  }
}

// ---------------------------------------------------------------------------
// GEMM1: qkv[row,n] = sum_k q[row,k] * w_qkv[n,k]  (bf16 operands, f32 accum)
//   n < 8192 -> qk bf16 (qs|ks); n >= 8192 -> vs fp32.
// ---------------------------------------------------------------------------
__global__ void __launch_bounds__(TB)
k_gemm_qkv(const unsigned short* __restrict__ qbf, const unsigned short* __restrict__ wbf,
           unsigned short* __restrict__ qk, float* __restrict__ vs)
{
  const int M = kBS, N = kDTOT, K = kDQ;
  __shared__ __align__(16) unsigned short As[2 * BM * LDT];
  __shared__ __align__(16) unsigned short Bs[2 * BM * LDT];

  const int mBase = blockIdx.y * BM;
  const int nBase = blockIdx.x * BN;

  v8f acc[2][4];
  #pragma unroll
  for (int i = 0; i < 2; ++i)
    #pragma unroll
    for (int j = 0; j < 4; ++j)
      #pragma unroll
      for (int r = 0; r < 8; ++r) acc[i][j][r] = 0.0f;

  wmma_mainloop(qbf + (size_t)mBase * K, wbf + (size_t)nBase * K,
                M - mBase, N - nBase, K, K, K, As, Bs, acc);

  const int lane = threadIdx.x & 31;
  const int wave = threadIdx.x >> 5;
  const int wm = wave >> 1, wn = wave & 1;
  const int h = lane >> 4, mn = lane & 15;

  #pragma unroll
  for (int i = 0; i < 2; ++i) {
    #pragma unroll
    for (int j = 0; j < 4; ++j) {
      const int gcol = nBase + wn * 64 + j * 16 + mn;
      if (gcol >= N) continue;
      const int growb = mBase + wm * 32 + i * 16 + 8 * h;
      #pragma unroll
      for (int r = 0; r < 8; ++r) {
        const int grow = growb + r;
        if (grow >= M) continue;
        const float val = acc[i][j][r];
        if (gcol < kLDQK)
          qk[(size_t)grow * kLDQK + gcol] = bf1(val);
        else
          vs[(size_t)grow * 8 + (gcol - kLDQK)] = val;
      }
    }
  }
}

// ---------------------------------------------------------------------------
// GEMM2: at[b,s,t] = (1/64)*dot(qs[b,s,:], ks[b,t,:]) + mask[s,t]
// ---------------------------------------------------------------------------
__global__ void __launch_bounds__(TB)
k_gemm_attn(const unsigned short* __restrict__ qk, const float* __restrict__ mask,
            float* __restrict__ at)
{
  const int M = kS, N = kS, K = kDQ;
  const float scale = 0.015625f;   // 1/sqrt(4096)
  __shared__ __align__(16) unsigned short As[2 * BM * LDT];
  __shared__ __align__(16) unsigned short Bs[2 * BM * LDT];

  const int b = blockIdx.z;
  const unsigned short* Ab = qk + (size_t)b * kS * kLDQK;   // qs rows
  const unsigned short* Bb = Ab + kDQ;                      // ks rows

  const int mBase = blockIdx.y * BM;
  const int nBase = blockIdx.x * BN;

  v8f acc[2][4];
  #pragma unroll
  for (int i = 0; i < 2; ++i)
    #pragma unroll
    for (int j = 0; j < 4; ++j)
      #pragma unroll
      for (int r = 0; r < 8; ++r) acc[i][j][r] = 0.0f;

  wmma_mainloop(Ab + (size_t)mBase * kLDQK, Bb + (size_t)nBase * kLDQK,
                M - mBase, N - nBase, kLDQK, kLDQK, K, As, Bs, acc);

  const int lane = threadIdx.x & 31;
  const int wave = threadIdx.x >> 5;
  const int wm = wave >> 1, wn = wave & 1;
  const int h = lane >> 4, mn = lane & 15;

  float* atb = at + (size_t)b * kS * kS;
  #pragma unroll
  for (int i = 0; i < 2; ++i) {
    #pragma unroll
    for (int j = 0; j < 4; ++j) {
      const int gcol = nBase + wn * 64 + j * 16 + mn;
      if (gcol >= N) continue;
      const int growb = mBase + wm * 32 + i * 16 + 8 * h;
      #pragma unroll
      for (int r = 0; r < 8; ++r) {
        const int grow = growb + r;
        if (grow >= M) continue;
        atb[(size_t)grow * kS + gcol] =
            acc[i][j][r] * scale + mask[(size_t)grow * kS + gcol];
      }
    }
  }
}

// ---------------------------------------------------------------------------
// Row normalization: at /= max(rowsum, 0); then where(mask!=0) -> -1e9
// ---------------------------------------------------------------------------
__global__ void __launch_bounds__(TB)
k_norm_rows(float* __restrict__ at, const float* __restrict__ mask)
{
  const int b = blockIdx.x / kS;
  const int s = blockIdx.x % kS;
  float* row = at + ((size_t)b * kS + s) * kS;
  const float* mrow = mask + (size_t)s * kS;
  __shared__ float red[TB];

  float sum = 0.0f;
  for (int t = threadIdx.x; t < kS; t += TB) sum += row[t];
  red[threadIdx.x] = sum;
  __syncthreads();
  #pragma unroll
  for (int off = TB / 2; off > 0; off >>= 1) {
    if (threadIdx.x < off) red[threadIdx.x] += red[threadIdx.x + off];
    __syncthreads();
  }
  const float denom = fmaxf(red[0], 0.0f);
  for (int t = threadIdx.x; t < kS; t += TB) {
    const float v = row[t] / denom;               // div-by-0 -> inf, as in ref
    row[t] = (mrow[t] != 0.0f) ? -1.0e9f : v;
  }
}

// ---------------------------------------------------------------------------
// v_out[b,s,v] = sum_t at[b,s,t] * vs[b,t,v]   (DV = 8)
// ---------------------------------------------------------------------------
__global__ void __launch_bounds__(TB)
k_attn_v(const float* __restrict__ at, const float* __restrict__ vs,
         float* __restrict__ vout)
{
  const int b = blockIdx.x / kS;
  const int s = blockIdx.x % kS;
  const float* arow = at + ((size_t)b * kS + s) * kS;
  const float* vb   = vs + (size_t)b * kS * 8;

  float acc[8] = {0, 0, 0, 0, 0, 0, 0, 0};
  for (int t = threadIdx.x; t < kS; t += TB) {
    const float a = arow[t];
    const float4 v0 = *(const float4*)(vb + (size_t)t * 8);
    const float4 v1 = *(const float4*)(vb + (size_t)t * 8 + 4);
    acc[0] += a * v0.x; acc[1] += a * v0.y; acc[2] += a * v0.z; acc[3] += a * v0.w;
    acc[4] += a * v1.x; acc[5] += a * v1.y; acc[6] += a * v1.z; acc[7] += a * v1.w;
  }
  __shared__ float red[TB * 8];
  #pragma unroll
  for (int j = 0; j < 8; ++j) red[threadIdx.x * 8 + j] = acc[j];
  __syncthreads();
  for (int off = TB / 2; off > 0; off >>= 1) {
    if (threadIdx.x < off) {
      #pragma unroll
      for (int j = 0; j < 8; ++j)
        red[threadIdx.x * 8 + j] += red[(threadIdx.x + off) * 8 + j];
    }
    __syncthreads();
  }
  if (threadIdx.x < 8)
    vout[((size_t)b * kS + s) * 8 + threadIdx.x] = red[threadIdx.x];
}

// ---------------------------------------------------------------------------
// out[b,o] = sum_j vout[b,j] * w_proj[o,j]
// ---------------------------------------------------------------------------
__global__ void __launch_bounds__(TB)
k_proj(const float* __restrict__ vout, const float* __restrict__ w_proj,
       float* __restrict__ out)
{
  const int o = blockIdx.x;
  const float* wrow = w_proj + (size_t)o * kDTOT;

  float acc[8] = {0, 0, 0, 0, 0, 0, 0, 0};
  for (int j = threadIdx.x; j < kDTOT; j += TB) {
    const float w = wrow[j];
    #pragma unroll
    for (int b = 0; b < 8; ++b) acc[b] += w * vout[(size_t)b * kDTOT + j];
  }
  __shared__ float red[TB * 8];
  #pragma unroll
  for (int b = 0; b < 8; ++b) red[threadIdx.x * 8 + b] = acc[b];
  __syncthreads();
  for (int off = TB / 2; off > 0; off >>= 1) {
    if (threadIdx.x < off) {
      #pragma unroll
      for (int b = 0; b < 8; ++b)
        red[threadIdx.x * 8 + b] += red[(threadIdx.x + off) * 8 + b];
    }
    __syncthreads();
  }
  if (threadIdx.x < 8)
    out[(size_t)threadIdx.x * kDO + o] = red[threadIdx.x];
}

// ---------------------------------------------------------------------------
extern "C" void kernel_launch(void* const* d_in, const int* in_sizes, int n_in,
                              void* d_out, int out_size, void* d_ws, size_t ws_size,
                              hipStream_t stream)
{
  (void)in_sizes; (void)n_in; (void)out_size; (void)ws_size;
  const float* q      = (const float*)d_in[0];
  // d_in[1] (k) and d_in[2] (v) are unused by the reference computation.
  const float* w_qkv  = (const float*)d_in[3];
  const float* w_proj = (const float*)d_in[4];
  const float* mask   = (const float*)d_in[5];
  float* out = (float*)d_out;

  char* ws = (char*)d_ws;
  const size_t sz_qbf = (size_t)kBS * kDQ * sizeof(unsigned short);   //  67,174,400
  const size_t sz_wbf = (size_t)kDTOT * kDQ * sizeof(unsigned short); //  67,174,400
  const size_t sz_qk  = (size_t)kBS * kLDQK * sizeof(unsigned short); // 134,348,800
  unsigned short* qbf = (unsigned short*)(ws);
  unsigned short* wbf = (unsigned short*)(ws + sz_qbf);
  unsigned short* qk  = (unsigned short*)(ws + sz_qbf + sz_wbf);
  float* vs   = (float*)(ws + sz_qbf + sz_wbf + sz_qk);
  // qbf/wbf are dead after GEMM1 -> reuse their space for at / vout
  float* at   = (float*)(ws);                // 33.6 MB < 67 MB (qbf region)
  float* vout = (float*)(ws + sz_qbf);       // in wbf region

  dim3 blk(TB);
  k_cvt_bf16<<<dim3(2048), blk, 0, stream>>>(q,     qbf, (kBS * kDQ) / 4);
  k_cvt_bf16<<<dim3(2048), blk, 0, stream>>>(w_qkv, wbf, (kDTOT * kDQ) / 4);

  dim3 g1((kDTOT + BN - 1) / BN, (kBS + BM - 1) / BM);      // 65 x 65
  k_gemm_qkv<<<g1, blk, 0, stream>>>(qbf, wbf, qk, vs);

  dim3 g2((kS + BN - 1) / BN, (kS + BM - 1) / BM, kB);      // 9 x 9 x 8
  k_gemm_attn<<<g2, blk, 0, stream>>>(qk, mask, at);

  k_norm_rows<<<dim3(kBS), blk, 0, stream>>>(at, mask);
  k_attn_v<<<dim3(kBS), blk, 0, stream>>>(at, vs, vout);
  k_proj<<<dim3(kDO), blk, 0, stream>>>(vout, w_proj, out);
}
